// SAVSSM_23673859735924
// MI455X (gfx1250) — compile-verified, matrix operand order checked by reading, not run on hardware
//
#include <hip/hip_runtime.h>
#include <hip/hip_bf16.h>

// Problem constants (from the reference)
#define B_    4
#define C_    64
#define H_    64
#define W_    64
#define L_    4096      // H*W
#define R_    64
#define D_    128       // 2*C
#define N_    16
#define K_    4
#define DTR_  4         // (C+15)/16
#define TWOD_ 256
#define XPC_  36        // DT_RANK + 2N
#define EPSF  1e-5f
#define MT_   4         // M-tiles per wave in WMMA kernels (B-fragment reuse)

typedef __attribute__((ext_vector_type(16))) __bf16 bf16x16;
typedef __attribute__((ext_vector_type(8)))  float  f32x8;

// scan-direction source index: xs[b,k,d,l] = xc[b,d,sidx(k,l)]
__device__ __forceinline__ int sidx(int k, int l) {
  switch (k) {
    case 0:  return l;
    case 1:  return ((l & 63) << 6) | (l >> 6);          // (l%W)*H + l/W
    case 2:  return L_ - 1 - l;
    default: { int m = L_ - 1 - l; return ((m & 63) << 6) | (m >> 6); }
  }
}

__device__ __forceinline__ float sigmoidf_(float x) { return 1.f / (1.f + __expf(-x)); }
__device__ __forceinline__ float siluf_(float x)    { return x * sigmoidf_(x); }
__device__ __forceinline__ float softplusf_(float x) {
  return (x > 20.f) ? x : log1pf(__expf(x));
}

// ---------------------------------------------------------------------------
// 1) FiLM modulators + channel attention (all tiny GEMVs from representation)
// ---------------------------------------------------------------------------
__global__ void modulators_kernel(const float* __restrict__ rep,
                                  const float* __restrict__ Wg, const float* __restrict__ bg,
                                  const float* __restrict__ Wb, const float* __restrict__ bb,
                                  const float* __restrict__ Wc, const float* __restrict__ bc,
                                  const float* __restrict__ Wf1, const float* __restrict__ bf1,
                                  const float* __restrict__ Wf2, const float* __restrict__ bf2,
                                  float* __restrict__ gamma, float* __restrict__ beta,
                                  float* __restrict__ cond,  float* __restrict__ att) {
  const int b = blockIdx.x;
  const int t = threadIdx.x;
  const float* r = rep + b * R_;
  __shared__ float a16[C_ / 4];

  if (t < C_) {                       // gamma
    float s = bg[t];
    #pragma unroll 8
    for (int j = 0; j < R_; ++j) s += r[j] * Wg[t * R_ + j];
    gamma[b * C_ + t] = s;
  } else if (t < 2 * C_) {            // beta
    int c = t - C_;
    float s = bb[c];
    #pragma unroll 8
    for (int j = 0; j < R_; ++j) s += r[j] * Wb[c * R_ + j];
    beta[b * C_ + c] = s;
  } else {                            // cond = 1 + rep@Wc^T + bc
    int d = t - 2 * C_;
    float s = 1.f + bc[d];
    #pragma unroll 8
    for (int j = 0; j < R_; ++j) s += r[j] * Wc[d * R_ + j];
    cond[b * D_ + d] = s;
  }

  if (t < C_ / 4) {                   // fc1 + relu
    float s = bf1[t];
    #pragma unroll 8
    for (int j = 0; j < R_; ++j) s += r[j] * Wf1[t * R_ + j];
    a16[t] = fmaxf(s, 0.f);
  }
  __syncthreads();
  if (t < C_) {                       // fc2 + sigmoid
    float s = bf2[t];
    #pragma unroll
    for (int j = 0; j < C_ / 4; ++j) s += a16[j] * Wf2[t * (C_ / 4) + j];
    att[b * C_ + t] = sigmoidf_(s);
  }
}

// ---------------------------------------------------------------------------
// 2) Instance norm + FiLM, writes channels-last x_bhwc (B,L,C).
// ---------------------------------------------------------------------------
__global__ void instnorm_kernel(const float* __restrict__ input,
                                const float* __restrict__ gamma,
                                const float* __restrict__ beta,
                                float* __restrict__ xbhwc) {
  const int bc = blockIdx.x;              // b*C + c
  const int c  = bc % C_;
  const int b  = bc / C_;
  const int t  = threadIdx.x;
  const float* src = input + (size_t)bc * L_;

  float s = 0.f, ss = 0.f;
  for (int l = t; l < L_; l += 256) { float v = src[l]; s += v; ss += v * v; }
  __shared__ float r1[256], r2[256];
  r1[t] = s; r2[t] = ss; __syncthreads();
  for (int st = 128; st > 0; st >>= 1) {
    if (t < st) { r1[t] += r1[t + st]; r2[t] += r2[t + st]; }
    __syncthreads();
  }
  const float mu   = r1[0] * (1.f / L_);
  const float var  = r2[0] * (1.f / L_) - mu * mu;
  const float rstd = rsqrtf(var + EPSF);
  const float g    = 1.f + gamma[bc];
  const float bt   = beta[bc];
  for (int l = t; l < L_; l += 256)
    xbhwc[(size_t)(b * L_ + l) * C_ + c] = (src[l] - mu) * rstd * g + bt;
}

// ---------------------------------------------------------------------------
// 3) xz = x_bhwc(BL,64) @ W_in^T(64,256) — bf16 WMMA.
//    One wave computes a 64x16 strip: 4 M-tiles share each B fragment.
// ---------------------------------------------------------------------------
__global__ void gemm_in_kernel(const float* __restrict__ X,
                               const float* __restrict__ Win,   // (2D, C) row-major
                               float* __restrict__ XZ) {
  const int lane = threadIdx.x;           // blockDim = 32
  const int nt   = blockIdx.x % (TWOD_ / 16);
  const int mt   = blockIdx.x / (TWOD_ / 16);
  const int m0 = mt * (16 * MT_), n0 = nt * 16;
  const int mrow = lane & 15, half = lane >> 4;

  f32x8 acc[MT_] = {};
  const float* brow = Win + (size_t)(n0 + mrow) * C_;   // B[k][n] = Win[n][k]
  #pragma unroll
  for (int kk0 = 0; kk0 < C_; kk0 += 32) {
    bf16x16 bm;
    #pragma unroll
    for (int e = 0; e < 8; ++e) {
      bm[e]     = (__bf16)brow[kk0 + half * 8 + e];
      bm[8 + e] = (__bf16)brow[kk0 + 16 + half * 8 + e];
    }
    #pragma unroll
    for (int t = 0; t < MT_; ++t) {
      const float* arow = X + (size_t)(m0 + t * 16 + mrow) * C_;
      bf16x16 a;
      #pragma unroll
      for (int e = 0; e < 8; ++e) {
        a[e]     = (__bf16)arow[kk0 + half * 8 + e];
        a[8 + e] = (__bf16)arow[kk0 + 16 + half * 8 + e];
      }
      acc[t] = __builtin_amdgcn_wmma_f32_16x16x32_bf16(false, a, false, bm,
                                                       (short)0, acc[t], false, false);
    }
  }
  const int col = n0 + (lane & 15);
  #pragma unroll
  for (int t = 0; t < MT_; ++t) {
    const int rbase = m0 + t * 16 + half * 8;
    #pragma unroll
    for (int i = 0; i < 8; ++i)
      XZ[(size_t)(rbase + i) * TWOD_ + col] = acc[t][i];
  }
}

// ---------------------------------------------------------------------------
// 4) Depthwise 3x3 SAME conv + bias + SiLU → xc (B,D,L) channels-first.
// ---------------------------------------------------------------------------
__global__ void dwconv_kernel(const float* __restrict__ XZ,
                              const float* __restrict__ cw,   // (D,1,3,3)
                              const float* __restrict__ cb,
                              float* __restrict__ XC) {
  const int t = blockIdx.x * 256 + threadIdx.x;   // over B*D*L, (b,d,l)
  const int l  = t & (L_ - 1);
  const int bd = t >> 12;
  const int d  = bd % D_;
  const int b  = bd / D_;
  const int h = l >> 6, w = l & 63;

  float s = cb[d];
  #pragma unroll
  for (int i = 0; i < 3; ++i) {
    const int hh = h + i - 1;
    if (hh < 0 || hh >= H_) continue;
    #pragma unroll
    for (int j = 0; j < 3; ++j) {
      const int ww = w + j - 1;
      if (ww < 0 || ww >= W_) continue;
      s += XZ[(size_t)(b * L_ + hh * W_ + ww) * TWOD_ + d] * cw[d * 9 + i * 3 + j];
    }
  }
  XC[t] = siluf_(s);
}

// ---------------------------------------------------------------------------
// 5) x_dbl[b,k,c,l] = sum_d xc[b,d,sidx(k,l)] * x_proj_w[k,c,d]
//    bf16 WMMA: wave computes 64 l-rows x 16 c-cols; B (weights) reused 4x.
// ---------------------------------------------------------------------------
__global__ void xdbl_gemm_kernel(const float* __restrict__ XC,
                                 const float* __restrict__ XPW,  // (K,36,D)
                                 float* __restrict__ XDBL) {     // (B,K,36,L)
  const int lane = threadIdx.x;
  int idx = blockIdx.x;
  const int ct = idx % 3;  idx /= 3;
  const int lt = idx % (L_ / (16 * MT_)); idx /= (L_ / (16 * MT_));
  const int k  = idx % K_;
  const int b  = idx / K_;
  const int l0 = lt * (16 * MT_), c0 = ct * 16;
  const int mrow = lane & 15, half = lane >> 4;

  int lsrc[MT_];
  #pragma unroll
  for (int t = 0; t < MT_; ++t) lsrc[t] = sidx(k, l0 + t * 16 + mrow);
  const float* xcb = XC + (size_t)b * D_ * L_;
  const int crow = c0 + mrow;                     // N index for B fragment

  f32x8 acc[MT_] = {};
  #pragma unroll
  for (int kk0 = 0; kk0 < D_; kk0 += 32) {
    bf16x16 bm;
    #pragma unroll
    for (int e = 0; e < 8; ++e) {
      const int kkA = kk0 + half * 8 + e;
      const int kkB = kk0 + 16 + half * 8 + e;
      bm[e]     = (crow < XPC_) ? (__bf16)XPW[((size_t)k * XPC_ + crow) * D_ + kkA] : (__bf16)0.f;
      bm[8 + e] = (crow < XPC_) ? (__bf16)XPW[((size_t)k * XPC_ + crow) * D_ + kkB] : (__bf16)0.f;
    }
    #pragma unroll
    for (int t = 0; t < MT_; ++t) {
      bf16x16 a;
      #pragma unroll
      for (int e = 0; e < 8; ++e) {
        a[e]     = (__bf16)xcb[(size_t)(kk0 + half * 8 + e) * L_ + lsrc[t]];
        a[8 + e] = (__bf16)xcb[(size_t)(kk0 + 16 + half * 8 + e) * L_ + lsrc[t]];
      }
      acc[t] = __builtin_amdgcn_wmma_f32_16x16x32_bf16(false, a, false, bm,
                                                       (short)0, acc[t], false, false);
    }
  }
  const int col = c0 + (lane & 15);               // c index
  if (col < XPC_) {
    #pragma unroll
    for (int t = 0; t < MT_; ++t) {
      const int rbase = l0 + t * 16 + half * 8;   // l index
      #pragma unroll
      for (int i = 0; i < 8; ++i)
        XDBL[((size_t)(b * K_ + k) * XPC_ + col) * L_ + rbase + i] = acc[t][i];
    }
  }
}

// ---------------------------------------------------------------------------
// 6) dt[b,k,d,l] = softplus( sum_r x_dbl[b,k,r,l]*dt_w[k,d,r] + dt_b[k,d] )
// ---------------------------------------------------------------------------
__global__ void dt_kernel(const float* __restrict__ XDBL,
                          const float* __restrict__ DTW,  // (K,D,DTR)
                          const float* __restrict__ DTB,  // (K,D)
                          float* __restrict__ DT) {
  const int t = blockIdx.x * 256 + threadIdx.x;   // over B*K*D*L, (b,k,d,l)
  const int l   = t & (L_ - 1);
  int bkd = t >> 12;
  const int d = bkd % D_; bkd /= D_;
  const int k = bkd % K_;
  const int b = bkd / K_;
  float acc = DTB[k * D_ + d];
  #pragma unroll
  for (int r = 0; r < DTR_; ++r)
    acc += XDBL[((size_t)(b * K_ + k) * XPC_ + r) * L_ + l] * DTW[(k * D_ + d) * DTR_ + r];
  DT[t] = softplusf_(acc);
}

// ---------------------------------------------------------------------------
// 7) Selective scan. 16 lanes (= states n) per (b,k,d) sequence.
// ---------------------------------------------------------------------------
__global__ void scan_kernel(const float* __restrict__ XC,
                            const float* __restrict__ DT,
                            const float* __restrict__ XDBL,
                            const float* __restrict__ Alogs,  // (K*D, N)
                            const float* __restrict__ Ds,     // (K*D)
                            float* __restrict__ YS) {
  const int g = blockIdx.x * 8 + (threadIdx.x >> 4);   // group = (b,k,d)
  const int n = threadIdx.x & 15;
  if (g >= B_ * K_ * D_) return;
  const int d = g % D_;
  const int k = (g / D_) % K_;
  const int b = g / (K_ * D_);

  const float An   = -__expf(Alogs[(k * D_ + d) * N_ + n]);
  const float Dval = Ds[k * D_ + d];

  const float* dtp = DT + (size_t)g * L_;
  const float* xcp = XC + (size_t)(b * D_ + d) * L_;
  const float* Bp  = XDBL + ((size_t)(b * K_ + k) * XPC_ + DTR_ + n) * L_;
  const float* Cp  = XDBL + ((size_t)(b * K_ + k) * XPC_ + DTR_ + N_ + n) * L_;
  float* ysp = YS + (size_t)g * L_;

  float h = 0.f;
  for (int l = 0; l < L_; ++l) {
    __builtin_prefetch(dtp + l + 16, 0, 0);
    __builtin_prefetch(Bp + l + 16, 0, 0);
    __builtin_prefetch(Cp + l + 16, 0, 0);
    const float dtv = dtp[l];
    const float xv  = xcp[sidx(k, l)];
    const float Bv  = Bp[l];
    const float Cv  = Cp[l];
    h = h * __expf(dtv * An) + (dtv * xv) * Bv;
    float y = h * Cv;
    y += __shfl_xor(y, 1);
    y += __shfl_xor(y, 2);
    y += __shfl_xor(y, 4);
    y += __shfl_xor(y, 8);
    if (n == 0) ysp[l] = y + xv * Dval;
  }
}

// ---------------------------------------------------------------------------
// 8) Merge 4 directions, LayerNorm over D, cond modulation, SiLU(z) gating.
// ---------------------------------------------------------------------------
__global__ void merge_ln_kernel(const float* __restrict__ YS,
                                const float* __restrict__ XZ,
                                const float* __restrict__ cond,
                                const float* __restrict__ lnw,
                                const float* __restrict__ lnb,
                                float* __restrict__ YG) {
  const int bl = blockIdx.x;            // b*L + l
  const int b = bl >> 12;
  const int l = bl & (L_ - 1);
  const int d = threadIdx.x;
  const int wh = sidx(1, l);

  float v = YS[((size_t)(b * K_ + 0) * D_ + d) * L_ + l]
          + YS[((size_t)(b * K_ + 2) * D_ + d) * L_ + (L_ - 1 - l)]
          + YS[((size_t)(b * K_ + 1) * D_ + d) * L_ + wh]
          + YS[((size_t)(b * K_ + 3) * D_ + d) * L_ + (L_ - 1 - wh)];

  __shared__ float r1[D_], r2[D_];
  r1[d] = v; r2[d] = v * v; __syncthreads();
  for (int st = 64; st > 0; st >>= 1) {
    if (d < st) { r1[d] += r1[d + st]; r2[d] += r2[d + st]; }
    __syncthreads();
  }
  const float mu  = r1[0] * (1.f / D_);
  const float var = r2[0] * (1.f / D_) - mu * mu;

  float y = (v - mu) * rsqrtf(var + EPSF) * lnw[d] + lnb[d];
  y *= cond[b * D_ + d];
  const float z = XZ[(size_t)bl * TWOD_ + D_ + d];
  y *= siluf_(z);
  YG[(size_t)bl * D_ + d] = y;
}

// ---------------------------------------------------------------------------
// 9) out_bld(BL,64) = ygated(BL,128) @ W_out^T — bf16 WMMA, 64x16 per wave.
// ---------------------------------------------------------------------------
__global__ void gemm_out_kernel(const float* __restrict__ YG,
                                const float* __restrict__ Wout,  // (C, D)
                                float* __restrict__ OBLD) {
  const int lane = threadIdx.x;
  const int nt = blockIdx.x % (C_ / 16);
  const int mt = blockIdx.x / (C_ / 16);
  const int m0 = mt * (16 * MT_), n0 = nt * 16;
  const int mrow = lane & 15, half = lane >> 4;

  f32x8 acc[MT_] = {};
  const float* brow = Wout + (size_t)(n0 + mrow) * D_;
  #pragma unroll
  for (int kk0 = 0; kk0 < D_; kk0 += 32) {
    bf16x16 bm;
    #pragma unroll
    for (int e = 0; e < 8; ++e) {
      bm[e]     = (__bf16)brow[kk0 + half * 8 + e];
      bm[8 + e] = (__bf16)brow[kk0 + 16 + half * 8 + e];
    }
    #pragma unroll
    for (int t = 0; t < MT_; ++t) {
      const float* arow = YG + (size_t)(m0 + t * 16 + mrow) * D_;
      bf16x16 a;
      #pragma unroll
      for (int e = 0; e < 8; ++e) {
        a[e]     = (__bf16)arow[kk0 + half * 8 + e];
        a[8 + e] = (__bf16)arow[kk0 + 16 + half * 8 + e];
      }
      acc[t] = __builtin_amdgcn_wmma_f32_16x16x32_bf16(false, a, false, bm,
                                                       (short)0, acc[t], false, false);
    }
  }
  const int col = n0 + (lane & 15);
  #pragma unroll
  for (int t = 0; t < MT_; ++t) {
    const int rbase = m0 + t * 16 + half * 8;
    #pragma unroll
    for (int i = 0; i < 8; ++i)
      OBLD[(size_t)(rbase + i) * C_ + col] = acc[t][i];
  }
}

// ---------------------------------------------------------------------------
// 10) out(B,C,H,W) = out_bld + input * att
// ---------------------------------------------------------------------------
__global__ void final_kernel(const float* __restrict__ OBLD,
                             const float* __restrict__ input,
                             const float* __restrict__ att,
                             float* __restrict__ out) {
  const int t = blockIdx.x * 256 + threadIdx.x;   // (b,c,l) flat = output layout
  const int l  = t & (L_ - 1);
  const int bc = t >> 12;
  const int c  = bc % C_;
  const int b  = bc / C_;
  out[t] = OBLD[(size_t)(b * L_ + l) * C_ + c] + input[t] * att[bc];
}

// ---------------------------------------------------------------------------
extern "C" void kernel_launch(void* const* d_in, const int* in_sizes, int n_in,
                              void* d_out, int out_size, void* d_ws, size_t ws_size,
                              hipStream_t stream) {
  const float* input = (const float*)d_in[0];
  const float* rep   = (const float*)d_in[1];
  const float* Wg    = (const float*)d_in[2];
  const float* bg    = (const float*)d_in[3];
  const float* Wb    = (const float*)d_in[4];
  const float* bb    = (const float*)d_in[5];
  const float* Win   = (const float*)d_in[6];
  const float* cw    = (const float*)d_in[7];
  const float* cb    = (const float*)d_in[8];
  const float* XPW   = (const float*)d_in[9];
  const float* DTW   = (const float*)d_in[10];
  const float* DTB   = (const float*)d_in[11];
  const float* Alogs = (const float*)d_in[12];
  const float* Ds    = (const float*)d_in[13];
  const float* lnw   = (const float*)d_in[14];
  const float* lnb   = (const float*)d_in[15];
  const float* Wc    = (const float*)d_in[16];
  const float* bc    = (const float*)d_in[17];
  const float* Wout  = (const float*)d_in[18];
  const float* Wf1   = (const float*)d_in[19];
  const float* bf1   = (const float*)d_in[20];
  const float* Wf2   = (const float*)d_in[21];
  const float* bf2   = (const float*)d_in[22];
  float* out = (float*)d_out;

  // Workspace layout (floats). dt buffer reused for ygated, x_bhwc for out_bld.
  float* ws = (float*)d_ws;
  size_t off = 0;
  float* gamma  = ws + off; off += B_ * C_;
  float* beta   = ws + off; off += B_ * C_;
  float* cond   = ws + off; off += B_ * D_;
  float* att    = ws + off; off += B_ * C_;
  off = (off + 511) & ~(size_t)511;
  float* xbhwc  = ws + off; off += (size_t)B_ * L_ * C_;       // 1.0M
  float* xz     = ws + off; off += (size_t)B_ * L_ * TWOD_;    // 4.2M
  float* xc     = ws + off; off += (size_t)B_ * D_ * L_;       // 2.1M
  float* xdbl   = ws + off; off += (size_t)B_ * K_ * XPC_ * L_;// 2.4M
  float* dt     = ws + off; off += (size_t)B_ * K_ * D_ * L_;  // 8.4M
  float* ys     = ws + off; off += (size_t)B_ * K_ * D_ * L_;  // 8.4M
  float* yg     = dt;       // dt dead after scan; merge_ln runs after scan
  float* obld   = xbhwc;    // xbhwc dead after gemm_in; reused for out GEMM
  (void)ws_size; (void)n_in; (void)in_sizes; (void)out_size;

  modulators_kernel<<<B_, 256, 0, stream>>>(rep, Wg, bg, Wb, bb, Wc, bc,
                                            Wf1, bf1, Wf2, bf2,
                                            gamma, beta, cond, att);
  instnorm_kernel<<<B_ * C_, 256, 0, stream>>>(input, gamma, beta, xbhwc);
  gemm_in_kernel<<<(B_ * L_ / (16 * MT_)) * (TWOD_ / 16), 32, 0, stream>>>(xbhwc, Win, xz);
  dwconv_kernel<<<(B_ * D_ * L_) / 256, 256, 0, stream>>>(xz, cw, cb, xc);
  xdbl_gemm_kernel<<<B_ * K_ * (L_ / (16 * MT_)) * 3, 32, 0, stream>>>(xc, XPW, xdbl);
  dt_kernel<<<(B_ * K_ * D_ * L_) / 256, 256, 0, stream>>>(xdbl, DTW, DTB, dt);
  scan_kernel<<<(B_ * K_ * D_ * 16) / 128, 128, 0, stream>>>(xc, dt, xdbl, Alogs, Ds, ys);
  merge_ln_kernel<<<B_ * L_, D_, 0, stream>>>(ys, xz, cond, lnw, lnb, yg);
  gemm_out_kernel<<<(B_ * L_ / (16 * MT_)) * (C_ / 16), 32, 0, stream>>>(yg, Wout, obld);
  final_kernel<<<(B_ * C_ * L_) / 256, 256, 0, stream>>>(obld, input, att, out);
}
